// Puzzler_87789131531026
// MI455X (gfx1250) — matrix operand order repaired and checked by reading, not compiled, not run on hardware
//
#include <hip/hip_runtime.h>
#include <hip/hip_bf16.h>
#include <math.h>

typedef __attribute__((ext_vector_type(16))) _Float16 v16h;
typedef __attribute__((ext_vector_type(8)))  _Float16 v8h;
typedef __attribute__((ext_vector_type(8)))  float    v8f;

#define KSZ   8
#define C_IN  3
#define HID   256
#define MID   256
#define NCLS  1000
#define NPIX  64          // K*K
#define HH    256
#define WW    256
#define BB    128
#define OUTV  1259        // 2+1+256+1000

// ---------------------------------------------------------------------------
// Dirichlet interpolation weights:  D(t) = e^{-i pi t} sin(pi H t)/sin(pi t)
// wc = cos(pi t)*S/H, ws = sin(pi t)*S/H  (S -> H cos(piHt)/cos(pit) at t in Z)
// ---------------------------------------------------------------------------
__device__ __forceinline__ void dirichlet256(float t, float& wc, float& ws) {
    float sp = sinpif(t);
    float cp = cospif(t);
    float sN = sinpif(256.0f * t);
    float cN = cospif(256.0f * t);
    float S  = (fabsf(sp) > 1e-5f) ? (sN / sp) : (256.0f * cN / cp);
    wc = cp * S * (1.0f / 256.0f);
    ws = sp * S * (1.0f / 256.0f);
}

// grid = B*64 blocks, 256 threads (thread = u index)
__global__ void weight_kernel(const float* __restrict__ loc,
                              const float* __restrict__ scl,
                              _Float16* __restrict__ A,     // [B][128][256] f16
                              float*    __restrict__ Wx)    // [B][2][64][256] f32
{
    int b = blockIdx.x >> 6;
    int n = blockIdx.x & 63;
    int u = threadIdx.x;
    int i = n >> 3, j = n & 7;
    float s = scl[b];
    float y = ((float)i * 0.125f - loc[b * 2 + 0]) * s;
    float x = ((float)j * 0.125f - loc[b * 2 + 1]) * s;
    float ty = y - (float)u * (1.0f / 256.0f);
    float tx = x - (float)u * (1.0f / 256.0f);
    float yc, ys, xc, xs;
    dirichlet256(ty, yc, ys);
    dirichlet256(tx, xc, xs);
    A[((size_t)(b * 128 + n)) * 256 + u]      = (_Float16)yc;  // cos rows 0..63
    A[((size_t)(b * 128 + n + 64)) * 256 + u] = (_Float16)ys;  // sin rows 64..127
    Wx[((size_t)((b * 2 + 0) * 64 + n)) * 256 + u] = xc;
    Wx[((size_t)((b * 2 + 1) * 64 + n)) * 256 + u] = xs;
}

// ---------------------------------------------------------------------------
// Per-(b,c,w-tile64) GEMM:  T[128x64] = A_b[128x256(K=u)] @ img[b,c][256x64]
// 4 waves, each owns a 16-col N slice, 8 M-tiles of 16x16x32 f16 WMMA.
// Epilogue: pix[b,n,c] += sum_w Tc[n,w]*wxc - Ts[n,w]*wxs  (atomic, W split 4x)
// ---------------------------------------------------------------------------
__global__ __launch_bounds__(128) void interp_gemm(
    const float*    __restrict__ img,   // [B][3][256][256]
    const _Float16* __restrict__ A,     // [B][128][256]
    const float*    __restrict__ Wx,    // [B][2][64][256]
    float*          __restrict__ pix)   // [B][64][3]
{
    __shared__ __align__(16) _Float16 ldsB[64 * 40];  // [w'][u'] transposed, pad 40

    int blk = blockIdx.x;
    int wt  = blk & 3;
    int c   = (blk >> 2) % 3;
    int b   = blk / 12;
    int tid  = threadIdx.x;
    int lane = tid & 31;
    int wave = tid >> 5;

    const float*    imgbc = img + ((size_t)(b * 3 + c)) * (HH * WW);
    const _Float16* Ab    = A + (size_t)b * 128 * 256;

    v8f zero = {0.f, 0.f, 0.f, 0.f, 0.f, 0.f, 0.f, 0.f};
    v8f acc[8];
#pragma unroll
    for (int m = 0; m < 8; ++m) acc[m] = zero;

    const int mrow   = lane & 15;
    const int koffA  = (lane < 16) ? 0 : 8;    // A: K 0..7 / 8..15 (and +16)
    const int kbB    = (lane < 16) ? 0 : 8;    // B: K 0..7 / 8..15 (and +16)
    const int wlocal = wave * 16 + (lane & 15);

    for (int kk = 0; kk < 8; ++kk) {
        int k0 = kk * 32;
        // stage image tile [32 u x 64 w] -> LDS transposed f16 [w][u]
        for (int r = 0; r < 16; ++r) {
            int idx = r * 128 + tid;
            int u = idx >> 6;          // 0..31
            int w = idx & 63;          // 0..63
            float v = imgbc[(size_t)(k0 + u) * WW + wt * 64 + w];
            ldsB[w * 40 + u] = (_Float16)v;
        }
        __syncthreads();

        // B fragment (shared by all 8 M-tiles this wave)
        const v8h* pb = (const v8h*)(&ldsB[wlocal * 40 + kbB]);
        v8h blo = pb[0];
        v8h bhi = pb[2];
        v16h fb = __builtin_shufflevector(blo, bhi, 0, 1, 2, 3, 4, 5, 6, 7,
                                          8, 9, 10, 11, 12, 13, 14, 15);
#pragma unroll
        for (int mt = 0; mt < 8; ++mt) {
            const _Float16* ap = Ab + (size_t)(mt * 16 + mrow) * 256 + k0 + koffA;
            v8h alo = *(const v8h*)(ap);
            v8h ahi = *(const v8h*)(ap + 16);
            v16h fa = __builtin_shufflevector(alo, ahi, 0, 1, 2, 3, 4, 5, 6, 7,
                                              8, 9, 10, 11, 12, 13, 14, 15);
            acc[mt] = __builtin_amdgcn_wmma_f32_16x16x32_f16(
                false, fa, false, fb, (short)0, acc[mt], false, false);
        }
        __syncthreads();
    }

    // Epilogue: pair cos tile (mt) with sin tile (mt+4); lane owns one w column.
    int wcol = wt * 64 + wlocal;
#pragma unroll
    for (int mt = 0; mt < 4; ++mt) {
#pragma unroll
        for (int i = 0; i < 8; ++i) {
            int n = mt * 16 + i + ((lane < 16) ? 0 : 8);
            float wc = Wx[((size_t)((b * 2 + 0) * 64 + n)) * 256 + wcol];
            float wsn = Wx[((size_t)((b * 2 + 1) * 64 + n)) * 256 + wcol];
            float v = acc[mt][i] * wc - acc[mt + 4][i] * wsn;
            v += __shfl_xor(v, 1);
            v += __shfl_xor(v, 2);
            v += __shfl_xor(v, 4);
            v += __shfl_xor(v, 8);
            if ((lane & 15) == 0)
                atomicAdd(&pix[(size_t)(b * 64 + n) * 3 + c], v);
        }
    }
}

// ---------------------------------------------------------------------------
// Whole MLP chain, one 256-thread block per batch row. ~0.2 GFLOP total.
// ---------------------------------------------------------------------------
__global__ __launch_bounds__(256) void mlp_kernel(
    const float* __restrict__ loc, const float* __restrict__ scl,
    const float* __restrict__ hid, const float* __restrict__ pix,
    const float* __restrict__ w0,  const float* __restrict__ b0,
    const float* __restrict__ rw,  const float* __restrict__ rb,
    const float* __restrict__ wf,  const float* __restrict__ bf,
    const float* __restrict__ cw0, const float* __restrict__ cb0,
    const float* __restrict__ crw, const float* __restrict__ crb,
    const float* __restrict__ cw1, const float* __restrict__ cb1,
    float* __restrict__ out)
{
    __shared__ float s_in[451];
    __shared__ float s_h[256];
    __shared__ float s_o[259];
    __shared__ float s_nh[256];
    __shared__ float s_c[256];
    __shared__ float s_c2[256];
    __shared__ float s_l[1000];
    __shared__ float s_red[256];

    int b = blockIdx.x, t = threadIdx.x;

    if (t < 2) s_in[t] = loc[b * 2 + t];
    if (t == 2) s_in[2] = scl[b];
    for (int j = t; j < 192; j += 256) s_in[3 + j] = pix[b * 192 + j];
    s_in[195 + t] = hid[b * 256 + t];
    __syncthreads();

    // h = inp @ w0 + b0
    {
        float a = b0[t];
        for (int k = 0; k < 451; ++k) a += s_in[k] * w0[k * 256 + t];
        s_h[t] = a;
    }
    __syncthreads();

    // 4 residual layers: h = relu(h@rw + rb + h)
    for (int L = 0; L < 4; ++L) {
        const float* W  = rw + (size_t)L * 256 * 256;
        const float* Bv = rb + L * 256;
        float a = Bv[t] + s_h[t];
        for (int k = 0; k < 256; ++k) a += s_h[k] * W[k * 256 + t];
        a = fmaxf(a, 0.0f);
        __syncthreads();
        s_h[t] = a;
        __syncthreads();
    }

    // out = tanh(h @ wf + bf)   (259)
    for (int j = t; j < 259; j += 256) {
        float a = bf[j];
        for (int k = 0; k < 256; ++k) a += s_h[k] * wf[k * 259 + j];
        s_o[j] = tanhf(a);
    }
    __syncthreads();

    if (t == 0) {
        out[(size_t)b * OUTV + 0] = loc[b * 2 + 0] + s_o[0];
        out[(size_t)b * OUTV + 1] = loc[b * 2 + 1] + s_o[1];
        out[(size_t)b * OUTV + 2] = scl[b] + s_o[2];
    }
    {
        float nh = hid[b * 256 + t] + s_o[3 + t];
        s_nh[t] = nh;
        out[(size_t)b * OUTV + 3 + t] = nh;
    }
    __syncthreads();

    // c = relu(nh @ cw0 + cb0)
    {
        float a = cb0[t];
        for (int k = 0; k < 256; ++k) a += s_nh[k] * cw0[k * 256 + t];
        s_c[t] = fmaxf(a, 0.0f);
    }
    __syncthreads();
    // c2 = relu(c @ crw + crb + c)
    {
        float a = crb[t] + s_c[t];
        for (int k = 0; k < 256; ++k) a += s_c[k] * crw[k * 256 + t];
        s_c2[t] = fmaxf(a, 0.0f);
    }
    __syncthreads();
    // logits
    for (int j = t; j < 1000; j += 256) {
        float a = cb1[j];
        for (int k = 0; k < 256; ++k) a += s_c2[k] * cw1[k * 1000 + j];
        s_l[j] = a;
    }
    __syncthreads();

    // softmax
    float m = -1e30f;
    for (int j = t; j < 1000; j += 256) m = fmaxf(m, s_l[j]);
    s_red[t] = m;
    __syncthreads();
    for (int off = 128; off > 0; off >>= 1) {
        if (t < off) s_red[t] = fmaxf(s_red[t], s_red[t + off]);
        __syncthreads();
    }
    float mx = s_red[0];
    __syncthreads();
    float sum = 0.0f;
    for (int j = t; j < 1000; j += 256) {
        float e = __expf(s_l[j] - mx);
        s_l[j] = e;
        sum += e;
    }
    s_red[t] = sum;
    __syncthreads();
    for (int off = 128; off > 0; off >>= 1) {
        if (t < off) s_red[t] += s_red[t + off];
        __syncthreads();
    }
    float inv = 1.0f / s_red[0];
    for (int j = t; j < 1000; j += 256)
        out[(size_t)b * OUTV + 259 + j] = s_l[j] * inv;
}

// ---------------------------------------------------------------------------
extern "C" void kernel_launch(void* const* d_in, const int* in_sizes, int n_in,
                              void* d_out, int out_size, void* d_ws, size_t ws_size,
                              hipStream_t stream) {
    const float* image = (const float*)d_in[0];
    const float* locations = (const float*)d_in[1];
    const float* scales = (const float*)d_in[2];
    const float* hidden = (const float*)d_in[3];
    const float* w0  = (const float*)d_in[4];
    const float* b0  = (const float*)d_in[5];
    const float* rw  = (const float*)d_in[6];
    const float* rb  = (const float*)d_in[7];
    const float* wf  = (const float*)d_in[8];
    const float* bf  = (const float*)d_in[9];
    const float* cw0 = (const float*)d_in[10];
    const float* cb0 = (const float*)d_in[11];
    const float* crw = (const float*)d_in[12];
    const float* crb = (const float*)d_in[13];
    const float* cw1 = (const float*)d_in[14];
    const float* cb1 = (const float*)d_in[15];

    char* ws = (char*)d_ws;
    const size_t szA  = (size_t)BB * 128 * 256 * sizeof(_Float16);  // 8 MiB
    const size_t szWx = (size_t)BB * 2 * 64 * 256 * sizeof(float);  // 16 MiB
    _Float16* A  = (_Float16*)ws;
    float*    Wx = (float*)(ws + szA);
    float*    pix = (float*)(ws + szA + szWx);                      // [B][64][3]

    hipMemsetAsync(pix, 0, (size_t)BB * 64 * 3 * sizeof(float), stream);

    weight_kernel<<<BB * 64, 256, 0, stream>>>(locations, scales, A, Wx);
    interp_gemm<<<BB * 3 * 4, 128, 0, stream>>>(image, A, Wx, pix);
    mlp_kernel<<<BB, 256, 0, stream>>>(locations, scales, hidden, pix,
                                       w0, b0, rw, rb, wf, bf,
                                       cw0, cb0, crw, crb, cw1, cb1,
                                       (float*)d_out);
}
